// StackedMatchLSTM_64072322121774
// MI455X (gfx1250) — compile-verified, matrix operand order checked by reading, not run on hardware
//
#include <hip/hip_runtime.h>
#include <hip/hip_bf16.h>

// ---------------------------------------------------------------------------
// Problem constants (from reference)
// ---------------------------------------------------------------------------
#define B   32
#define TP  512
#define TQ  64
#define DP  256
#define DQ  256
#define H   256
#define AH  256
#define NL  2
#define NEGV (-1000000000.0f)

typedef __attribute__((ext_vector_type(16))) __bf16 v16bf;
typedef __attribute__((ext_vector_type(8)))  __bf16 v8bf;
typedef __attribute__((ext_vector_type(4)))  __bf16 v4bf;
typedef __attribute__((ext_vector_type(8)))  float  v8f;
typedef __attribute__((ext_vector_type(4)))  int    v4i;

// Async global->LDS staging (CDNA5): guarded so the file still compiles if
// this toolchain doesn't declare the builtins (falls back to sync load/store).
// Probe result (round 2): builtin exists, params are v4i* (b128 payload).
#if defined(__gfx1250__) && __has_builtin(__builtin_amdgcn_global_load_async_to_lds_b128)
#define USE_ASYNC 1
typedef __attribute__((address_space(1))) v4i glob_v4i;
typedef __attribute__((address_space(3))) v4i lds_v4i;
#if __has_builtin(__builtin_amdgcn_s_wait_asynccnt)
#define WAIT_ASYNC() __builtin_amdgcn_s_wait_asynccnt(0)
#else
#define WAIT_ASYNC() asm volatile("s_wait_asynccnt 0x0" ::: "memory")
#endif
#else
#define USE_ASYNC 0
#define WAIT_ASYNC() ((void)0)
#endif

// ---------------------------------------------------------------------------
// Casts
// ---------------------------------------------------------------------------
__global__ void cast_f32_bf16(const float* __restrict__ in,
                              __bf16* __restrict__ out, int n) {
    int i = blockIdx.x * 256 + threadIdx.x;
    if (i < n) out[i] = (__bf16)in[i];
}

// in: (Kd x Nd) f32 row-major  ->  out: (Nd x Kd) bf16 row-major
__global__ void cast_transpose_f32_bf16(const float* __restrict__ in,
                                        __bf16* __restrict__ out,
                                        int Kd, int Nd) {
    int i = blockIdx.x * 256 + threadIdx.x;
    if (i < Kd * Nd) {
        int k = i / Nd, n = i - k * Nd;
        out[(size_t)n * Kd + k] = (__bf16)in[i];
    }
}

// ---------------------------------------------------------------------------
// bf16 GEMM, f32 accumulate: C[M,N] = A[M,K] @ BT[N,K]^T (+ bias).
// Block 256 threads = 8 waves; block tile 128x64; wave tile 32x32 -> 4 WMMA.
// Both tiles staged K-major [rows][32] in LDS (double buffered); every
// fragment is two aligned ds_load_b128. Async global->LDS when available.
// ---------------------------------------------------------------------------
#define BM 128
#define BN 64
#define BK 32

__device__ __forceinline__ v16bf ld_frag(const __bf16* row, int kh) {
    v8bf lo = *(const v8bf*)(row + kh);        // K kh..kh+7
    v8bf hi = *(const v8bf*)(row + kh + 16);   // K kh+16..kh+23
    return __builtin_shufflevector(lo, hi, 0, 1, 2, 3, 4, 5, 6, 7,
                                           8, 9, 10, 11, 12, 13, 14, 15);
}

__device__ __forceinline__ void stage16(const __bf16* g, __bf16* l) {
#if USE_ASYNC
    __builtin_amdgcn_global_load_async_to_lds_b128(
        (glob_v4i*)g, (lds_v4i*)l, 0, 0);
#else
    *(v8bf*)l = *(const v8bf*)g;
#endif
}

__global__ void wmma_gemm_bf16(const __bf16* __restrict__ A,   // M x K
                               const __bf16* __restrict__ BT,  // N x K
                               float* __restrict__ C,
                               int M, int N, int K,
                               const float* __restrict__ bias_f,
                               const float* __restrict__ bias_iog,
                               int Hb) {
    __shared__ __attribute__((aligned(16))) __bf16 As[2][BM][BK];
    __shared__ __attribute__((aligned(16))) __bf16 Bs[2][BN][BK];

    const int tid  = threadIdx.x;
    const int w    = tid >> 5;        // wave 0..7 (wave32)
    const int lane = tid & 31;
    const int wm   = w & 3;           // 4 row groups of 32
    const int wn   = w >> 2;          // 2 col groups of 32
    const int bm   = blockIdx.x * BM;
    const int bn   = blockIdx.y * BN;

    // stage one K-chunk: As = 512 x 16B chunks (2/thread), Bs = 256 (1/thread)
    auto stage = [&](int buf, int k0) {
        #pragma unroll
        for (int r = 0; r < 2; ++r) {
            int c = tid + r * 256;             // 0..511
            int m = c >> 2, part = (c & 3) * 8;
            stage16(A + (size_t)(bm + m) * K + k0 + part, &As[buf][m][part]);
        }
        {
            int m = tid >> 2, part = (tid & 3) * 8;
            stage16(BT + (size_t)(bn + m) * K + k0 + part, &Bs[buf][m][part]);
        }
    };

    v8f acc00 = {}, acc01 = {}, acc10 = {}, acc11 = {};

    stage(0, 0);
    WAIT_ASYNC();
    __syncthreads();

    int buf = 0;
    for (int k0 = 0; k0 < K; k0 += BK) {
        if (k0 + BK < K) stage(buf ^ 1, k0 + BK);   // overlap next-tile copy

        const int kh  = (lane >> 4) * 8;
        const int mr0 = wm * 32 + (lane & 15);
        const int nc0 = wn * 32 + (lane & 15);
        v16bf a0 = ld_frag(&As[buf][mr0][0], kh);
        v16bf a1 = ld_frag(&As[buf][mr0 + 16][0], kh);
        v16bf b0 = ld_frag(&Bs[buf][nc0][0], kh);
        v16bf b1 = ld_frag(&Bs[buf][nc0 + 16][0], kh);

        acc00 = __builtin_amdgcn_wmma_f32_16x16x32_bf16(
                    false, a0, false, b0, (short)0, acc00, false, false);
        acc01 = __builtin_amdgcn_wmma_f32_16x16x32_bf16(
                    false, a0, false, b1, (short)0, acc01, false, false);
        acc10 = __builtin_amdgcn_wmma_f32_16x16x32_bf16(
                    false, a1, false, b0, (short)0, acc10, false, false);
        acc11 = __builtin_amdgcn_wmma_f32_16x16x32_bf16(
                    false, a1, false, b1, (short)0, acc11, false, false);

        WAIT_ASYNC();
        __syncthreads();
        buf ^= 1;
    }

    // C/D layout: lanes 0-15 -> M=r, lanes 16-31 -> M=r+8; N = lane&15
    const int mb = bm + wm * 32 + ((lane >> 4) * 8);
    const int nb = bn + wn * 32 + (lane & 15);
    #pragma unroll
    for (int r = 0; r < 8; ++r) {
        int ms[2] = { mb + r, mb + 16 + r };
        int ns[2] = { nb, nb + 16 };
        float av[2][2] = { { acc00[r], acc01[r] }, { acc10[r], acc11[r] } };
        #pragma unroll
        for (int i = 0; i < 2; ++i)
            #pragma unroll
            for (int j = 0; j < 2; ++j) {
                int n = ns[j];
                float bv = 0.f;
                if (bias_f) bv = (n < Hb) ? bias_f[n] : bias_iog[n - Hb];
                C[(size_t)ms[i] * N + n] = av[i][j] + bv;
            }
    }
}

// ---------------------------------------------------------------------------
// Sequential Match-LSTM scan. One WG per batch element (grid=32), 256 thr.
// 320 KB LDS/WGP exploited: qWq[b] 64K + q[b] 64K + Wr(bf16) 128K + state,
// total 273,920 B. W_lstm[z,h] streamed from L2 as bf16.
// float region layout (floats):
//   0 qWq 16384 | 16384 q 16384 | 32768 h 256 | 33024 c 256 | 33280 u 256
//   33536 wa 256 | 33792 s 64 | 33856 alpha 64 | 33920 part 256
//   34176 v 512 | 34688 pre 1024  -> 35712 floats = 142848 B, then Wr bf16.
// ---------------------------------------------------------------------------
#define SMEM_FLOATS 35712
#define SMEM_SCAN_BYTES (SMEM_FLOATS * 4 + H * AH * 2)   // 273,920

__global__ void scan_kernel(const float* __restrict__ qWq,     // (B,TQ,AH)
                            const float* __restrict__ pWp,     // (B,TP,AH)
                            const float* __restrict__ xpre,    // (B,TP,4H)+bias
                            const float* __restrict__ q,       // (B,TQ,DQ)
                            const float* __restrict__ mask_p,  // (B,TP)
                            const float* __restrict__ mask_q,  // (B,TQ)
                            const __bf16* __restrict__ Wr_bf,  // (H,AH) bf16
                            const float* __restrict__ wa,      // (AH)
                            const __bf16* __restrict__ Wl_zh,  // (512,4H) bf16
                            float* __restrict__ x_out)         // (B,TP,H)
{
    extern __shared__ __attribute__((aligned(16))) char smem_raw[];
    float* smem    = (float*)smem_raw;
    float* sh_qWq  = smem;
    float* sh_q    = smem + 16384;
    float* sh_h    = smem + 32768;
    float* sh_c    = smem + 33024;
    float* sh_u    = smem + 33280;
    float* sh_wa   = smem + 33536;
    float* sh_s    = smem + 33792;
    float* sh_al   = smem + 33856;
    float* sh_part = smem + 33920;
    float* sh_v    = smem + 34176;   // [0,256)=z  [256,512)=h_old
    float* sh_pre  = smem + 34688;
    __bf16* sh_Wr  = (__bf16*)(smem_raw + SMEM_FLOATS * 4);

    const int b   = blockIdx.x;
    const int tid = threadIdx.x;

    // ---- preload per-batch working set + Wr into LDS ----
    for (int i = tid; i < TQ * AH; i += 256) sh_qWq[i] = qWq[(size_t)b * TQ * AH + i];
    for (int i = tid; i < TQ * DQ; i += 256) sh_q[i]   = q  [(size_t)b * TQ * DQ + i];
    {
        const v8bf* src = (const v8bf*)Wr_bf;
        v8bf*       dst = (v8bf*)sh_Wr;
        for (int i = tid; i < (H * AH) / 8; i += 256) dst[i] = src[i];
    }
    sh_wa[tid] = wa[tid];
    sh_h[tid] = 0.f;
    sh_c[tid] = 0.f;
    __syncthreads();

    const float* pWp_b  = pWp  + (size_t)b * TP * AH;
    const float* xpre_b = xpre + (size_t)b * TP * (4 * H);
    float*       out_b  = x_out + (size_t)b * TP * H;

    for (int t = 0; t < TP; ++t) {
        if (t + 1 < TP) {   // warm next step's rows (global_prefetch_b8)
            __builtin_prefetch(pWp_b  + (size_t)(t + 1) * AH + tid, 0, 1);
            __builtin_prefetch(xpre_b + (size_t)(t + 1) * (4 * H) + tid * 4, 0, 1);
        }

        // stage 1: u[a] = pWp[t,a] + sum_k h[k]*Wr[k,a]   (a = tid, Wr in LDS)
        {
            float acc = pWp_b[(size_t)t * AH + tid];
            #pragma unroll 8
            for (int k = 0; k < H; ++k)
                acc = fmaf(sh_h[k], (float)sh_Wr[k * AH + tid], acc);
            sh_u[tid] = acc;
        }
        __syncthreads();

        // stage 2: s partials — 4 threads per query position j
        {
            int j = tid >> 2, part = tid & 3;
            const float* row = sh_qWq + j * AH;
            float acc = 0.f;
            #pragma unroll 4
            for (int a = part * 64; a < part * 64 + 64; ++a)
                acc = fmaf(tanhf(row[a] + sh_u[a]), sh_wa[a], acc);
            sh_part[tid] = acc;
        }
        __syncthreads();

        // stage 3: finish s, apply query mask
        if (tid < TQ) {
            float sv = sh_part[tid * 4] + sh_part[tid * 4 + 1] +
                       sh_part[tid * 4 + 2] + sh_part[tid * 4 + 3];
            sh_s[tid] = (mask_q[(size_t)b * TQ + tid] > 0.f) ? sv : NEGV;
        }
        __syncthreads();

        // stage 4: softmax over TQ=64 in one wave32 (2 elems/lane)
        if (tid < 32) {
            float s0 = sh_s[tid], s1 = sh_s[tid + 32];
            float mx = fmaxf(s0, s1);
            #pragma unroll
            for (int off = 16; off > 0; off >>= 1)
                mx = fmaxf(mx, __shfl_xor(mx, off, 32));
            float e0 = expf(s0 - mx), e1 = expf(s1 - mx);
            float sum = e0 + e1;
            #pragma unroll
            for (int off = 16; off > 0; off >>= 1)
                sum += __shfl_xor(sum, off, 32);
            float inv = 1.f / sum;
            sh_al[tid]      = e0 * inv;
            sh_al[tid + 32] = e1 * inv;
        }
        __syncthreads();

        // stage 5: z[dq] = sum_j alpha[j]*q[j,dq]; v = [z, h_old]
        {
            float acc = 0.f;
            #pragma unroll 8
            for (int j = 0; j < TQ; ++j)
                acc = fmaf(sh_al[j], sh_q[j * DQ + tid], acc);
            sh_v[tid]       = acc;
            sh_v[256 + tid] = sh_h[tid];
        }
        __syncthreads();

        // stage 6: pre[n] = xpre[t,n] + sum_{k<512} v[k]*Wl_zh[k,n]  (bf16 W)
        {
            const float4* xp4 = (const float4*)(xpre_b + (size_t)t * (4 * H));
            float4 p = xp4[tid];
            for (int k = 0; k < DQ + H; ++k) {
                v4bf wv = *(const v4bf*)(Wl_zh + (size_t)k * (4 * H) + tid * 4);
                float vk = sh_v[k];
                p.x = fmaf(vk, (float)wv[0], p.x);
                p.y = fmaf(vk, (float)wv[1], p.y);
                p.z = fmaf(vk, (float)wv[2], p.z);
                p.w = fmaf(vk, (float)wv[3], p.w);
            }
            ((float4*)sh_pre)[tid] = p;
        }
        __syncthreads();

        // stage 7: LSTM cell, masked update, emit h
        {
            float f = sh_pre[tid];
            float i = sh_pre[256 + tid];
            float o = sh_pre[512 + tid];
            float g = sh_pre[768 + tid];
            float sf = 1.f / (1.f + expf(-f));
            float si = 1.f / (1.f + expf(-i));
            float so = 1.f / (1.f + expf(-o));
            float c  = sh_c[tid];
            float h0 = sh_h[tid];
            float m  = mask_p[(size_t)b * TP + t];
            float c1 = sf * c + si * tanhf(g);
            c1 = c1 * m + c * (1.f - m);
            float h1 = so * tanhf(c1);
            h1 = h1 * m + h0 * (1.f - m);
            sh_c[tid] = c1;
            sh_h[tid] = h1;
            out_b[(size_t)t * H + tid] = h1;
        }
        __syncthreads();
    }
}

// ---------------------------------------------------------------------------
// Host orchestration (inputs in setup_inputs order).
// ---------------------------------------------------------------------------
static inline size_t align256(size_t x) { return (x + 255) & ~(size_t)255; }

extern "C" void kernel_launch(void* const* d_in, const int* in_sizes, int n_in,
                              void* d_out, int out_size, void* d_ws, size_t ws_size,
                              hipStream_t stream) {
    (void)in_sizes; (void)n_in; (void)out_size; (void)ws_size;

    const float* input_p  = (const float*)d_in[0];
    const float* mask_p   = (const float*)d_in[1];
    const float* input_q  = (const float*)d_in[2];
    const float* mask_q   = (const float*)d_in[3];
    const float* W_att_p  = (const float*)d_in[4];
    const float* W_att_q  = (const float*)d_in[5];
    const float* W_att_r  = (const float*)d_in[6];
    const float* w_att    = (const float*)d_in[7];
    const float* W_lstm   = (const float*)d_in[8];
    const float* bias_f   = (const float*)d_in[9];
    const float* bias_iog = (const float*)d_in[10];
    float* out = (float*)d_out;

    const size_t MXT = (size_t)B * TP;      // 16384
    const size_t MQ  = (size_t)B * TQ;      // 2048
    char* ws = (char*)d_ws;
    size_t off = 0;
    __bf16* x_bf     = (__bf16*)(ws + off); off = align256(off + MXT * DP * sizeof(__bf16));
    __bf16* q_bf     = (__bf16*)(ws + off); off = align256(off + MQ  * DQ * sizeof(__bf16));
    __bf16* WqT_bf   = (__bf16*)(ws + off); off = align256(off + (size_t)AH * DQ * sizeof(__bf16));
    __bf16* WpT_bf   = (__bf16*)(ws + off); off = align256(off + (size_t)AH * DP * sizeof(__bf16));
    __bf16* WlxT_bf  = (__bf16*)(ws + off); off = align256(off + (size_t)4 * H * DP * sizeof(__bf16));
    __bf16* Wr_bf    = (__bf16*)(ws + off); off = align256(off + (size_t)H * AH * sizeof(__bf16));
    __bf16* Wlzh_bf  = (__bf16*)(ws + off); off = align256(off + (size_t)(DQ + H) * 4 * H * sizeof(__bf16));
    float*  qWq      = (float*)(ws + off);  off = align256(off + MQ  * AH * sizeof(float));
    float*  pWp      = (float*)(ws + off);  off = align256(off + MXT * AH * sizeof(float));
    float*  xpre     = (float*)(ws + off);  off = align256(off + MXT * 4 * H * sizeof(float));
    float*  x_mid    = (float*)(ws + off);  off = align256(off + MXT * H * sizeof(float));

    (void)hipFuncSetAttribute((const void*)scan_kernel,
                              hipFuncAttributeMaxDynamicSharedMemorySize,
                              SMEM_SCAN_BYTES);

    // q -> bf16 once (layer-invariant)
    {
        int n = (int)(MQ * DQ);
        cast_f32_bf16<<<(n + 255) / 256, 256, 0, stream>>>(input_q, q_bf, n);
    }

    for (int d = 0; d < NL; ++d) {
        const float* x_src = (d == 0) ? input_p : x_mid;
        const float* Wq_d  = W_att_q + (size_t)d * DQ * AH;
        const float* Wp_d  = W_att_p + (size_t)d * DP * AH;
        const float* Wr_d  = W_att_r + (size_t)d * H * AH;
        const float* wa_d  = w_att   + (size_t)d * AH;
        const float* Wl_d  = W_lstm  + (size_t)d * (DP + DQ + H) * 4 * H;
        const float* Wlzh  = Wl_d + (size_t)DP * 4 * H;     // rows DP..DP+512
        const float* bf_d  = bias_f   + (size_t)d * H;
        const float* big_d = bias_iog + (size_t)d * 3 * H;
        float* scan_out    = (d == NL - 1) ? out : x_mid;

        // casts / transposes for this layer
        { int n = (int)(MXT * DP);
          cast_f32_bf16<<<(n + 255) / 256, 256, 0, stream>>>(x_src, x_bf, n); }
        { int n = DQ * AH;
          cast_transpose_f32_bf16<<<(n + 255) / 256, 256, 0, stream>>>(Wq_d, WqT_bf, DQ, AH); }
        { int n = DP * AH;
          cast_transpose_f32_bf16<<<(n + 255) / 256, 256, 0, stream>>>(Wp_d, WpT_bf, DP, AH); }
        { int n = DP * 4 * H;
          cast_transpose_f32_bf16<<<(n + 255) / 256, 256, 0, stream>>>(Wl_d, WlxT_bf, DP, 4 * H); }
        { int n = H * AH;
          cast_f32_bf16<<<(n + 255) / 256, 256, 0, stream>>>(Wr_d, Wr_bf, n); }
        { int n = (DQ + H) * 4 * H;
          cast_f32_bf16<<<(n + 255) / 256, 256, 0, stream>>>(Wlzh, Wlzh_bf, n); }

        // hoisted WMMA GEMMs
        wmma_gemm_bf16<<<dim3(MQ / BM, AH / BN), 256, 0, stream>>>(
            q_bf, WqT_bf, qWq, (int)MQ, AH, DQ, nullptr, nullptr, 0);
        wmma_gemm_bf16<<<dim3(MXT / BM, AH / BN), 256, 0, stream>>>(
            x_bf, WpT_bf, pWp, (int)MXT, AH, DP, nullptr, nullptr, 0);
        wmma_gemm_bf16<<<dim3(MXT / BM, (4 * H) / BN), 256, 0, stream>>>(
            x_bf, WlxT_bf, xpre, (int)MXT, 4 * H, DP, bf_d, big_d, H);

        // sequential scan: one WG per batch element, big LDS
        scan_kernel<<<B, 256, SMEM_SCAN_BYTES, stream>>>(
            qWq, pWp, xpre, input_q, mask_p, mask_q,
            Wr_bf, wa_d, Wlzh_bf, scan_out);
    }
}
